// MambaBlock_2705829396612
// MI455X (gfx1250) — matrix operand order repaired and checked by reading, not compiled, not run on hardware
//
#include <hip/hip_runtime.h>

// ---------------------------------------------------------------------------
// Mamba block forward for gfx1250 (MI455X). Pipeline:
//  1. transpose+convert W_in / W_out -> bf16 [N][K]
//  2. LayerNorm -> bf16 xn
//  3. WMMA bf16 GEMM (TDM double-buffered): xz = xn @ W_in
//  4. depthwise causal conv(4) + SiLU -> xc
//  5. xc @ W_xp -> (B,C,dt_raw) per row (33 outputs, block reduction)
//  6. selective scan: h = exp(dt*A)*h + dt*B*x ; y = <h,C> + D*x
//  7. gate: yg = bf16(y * silu(z))
//  8. WMMA bf16 GEMM + fused residual: out = yg @ W_out + x
// ---------------------------------------------------------------------------

#define BATCH   2
#define SEQ     2048
#define DMODEL  1024
#define DINNER  2048
#define DSTATE  16
#define DCONV   4
#define MROWS   (BATCH * SEQ)          // 4096
#define NXZ     (2 * DINNER)           // 4096

typedef __attribute__((ext_vector_type(16))) __bf16 v16bf;
typedef __attribute__((ext_vector_type(8)))  __bf16 v8bf;
typedef __attribute__((ext_vector_type(8)))  float  v8f;
typedef __attribute__((ext_vector_type(4)))  unsigned int v4u;
typedef __attribute__((ext_vector_type(8)))  int v8i;
typedef __attribute__((ext_vector_type(4)))  int v4i;

#if defined(__has_builtin)
#if __has_builtin(__builtin_amdgcn_tensor_load_to_lds) && \
    __has_builtin(__builtin_amdgcn_s_wait_tensorcnt)
#define USE_TDM 1
#endif
#endif

__device__ __forceinline__ unsigned short f2bf(float f) {
    unsigned int u = __float_as_uint(f);
    u += 0x7FFFu + ((u >> 16) & 1u);   // round-to-nearest-even
    return (unsigned short)(u >> 16);
}

// ---------------------------------------------------------------------------
// Kernel 1: transpose + convert f32 [K][N] -> bf16 [N][K]
// ---------------------------------------------------------------------------
__global__ void k_transpose_bf16(const float* __restrict__ in,
                                 unsigned short* __restrict__ out,
                                 int K, int N) {
    int idx = blockIdx.x * blockDim.x + threadIdx.x;
    int total = K * N;
    if (idx >= total) return;
    int n = idx / K;
    int k = idx - n * K;
    out[idx] = f2bf(in[(size_t)k * N + n]);
}

// ---------------------------------------------------------------------------
// Kernel 2: LayerNorm row (1024) -> bf16
// ---------------------------------------------------------------------------
__global__ __launch_bounds__(256) void k_layernorm_bf16(
    const float* __restrict__ x, const float* __restrict__ g,
    const float* __restrict__ b, unsigned short* __restrict__ out) {
    __shared__ float ws[8];
    __shared__ float s_mu, s_rstd;
    const int row = blockIdx.x, tid = threadIdx.x;
    const float* xr = x + (size_t)row * DMODEL;
    float v[4];
    float s = 0.f;
#pragma unroll
    for (int i = 0; i < 4; ++i) { v[i] = xr[tid + i * 256]; s += v[i]; }
#pragma unroll
    for (int o = 16; o > 0; o >>= 1) s += __shfl_down(s, o, 32);
    if ((tid & 31) == 0) ws[tid >> 5] = s;
    __syncthreads();
    if (tid < 8) {
        s = ws[tid];
#pragma unroll
        for (int o = 4; o > 0; o >>= 1) s += __shfl_down(s, o, 8);
        if (tid == 0) s_mu = s * (1.f / DMODEL);
    }
    __syncthreads();
    const float mu = s_mu;
    float vs = 0.f;
#pragma unroll
    for (int i = 0; i < 4; ++i) { float d = v[i] - mu; vs += d * d; }
#pragma unroll
    for (int o = 16; o > 0; o >>= 1) vs += __shfl_down(vs, o, 32);
    __syncthreads();
    if ((tid & 31) == 0) ws[tid >> 5] = vs;
    __syncthreads();
    if (tid < 8) {
        vs = ws[tid];
#pragma unroll
        for (int o = 4; o > 0; o >>= 1) vs += __shfl_down(vs, o, 8);
        if (tid == 0) s_rstd = rsqrtf(vs * (1.f / DMODEL) + 1e-5f);
    }
    __syncthreads();
    const float rstd = s_rstd;
    unsigned short* orow = out + (size_t)row * DMODEL;
#pragma unroll
    for (int i = 0; i < 4; ++i) {
        int c = tid + i * 256;
        orow[c] = f2bf((v[i] - mu) * rstd * g[c] + b[c]);
    }
}

// ---------------------------------------------------------------------------
// WMMA bf16 GEMM.  A: [M][K] bf16 row-major, Bt: [N][K] bf16 (B transposed),
// C: [M][N] f32, compile-time-optional residual.
// 256 threads = 8 waves; block tile 128x128; K slice 32; double-buffered LDS.
// Tile staging via Tensor Data Mover (TENSORcnt) when available.
// ---------------------------------------------------------------------------
#define GBM 128
#define GBN 128
#define GBK 32

#ifdef USE_TDM
// Issue one TDM 2D tile load: 128 rows x 32 bf16 from row-major [?][K] matrix.
__device__ __forceinline__ void tdm_tile_load(const unsigned short* gbase,
                                              unsigned ldsOff,
                                              unsigned rowStart, int K, int k0) {
    unsigned long long ga = (unsigned long long)(size_t)gbase +
        ((unsigned long long)rowStart * (unsigned)K + (unsigned)k0) * 2ull;
    v4u g0;
    g0.x = 1u;                                   // count=1, user descriptor
    g0.y = ldsOff;                               // lds_addr (bytes)
    g0.z = (unsigned)(ga & 0xFFFFFFFFu);         // global_addr[31:0]
    g0.w = (unsigned)((ga >> 32) & 0x1FFFFFFu)   // global_addr[56:32]
         | (2u << 30);                           // type=2 ("image")
    const unsigned td0 = (unsigned)K;            // tensor_dim0
    const unsigned td1 = 4096u;                  // tensor_dim1 (rows, ample)
    const unsigned st0 = (unsigned)K;            // tensor_dim0_stride
    v8i g1;
    g1[0] = (int)(1u << 16);                               // data_size=2B
    g1[1] = (int)((td0 & 0xFFFFu) << 16);                  // tensor_dim0 lo16
    g1[2] = (int)((td0 >> 16) | ((td1 & 0xFFFFu) << 16));  // td0 hi / td1 lo
    g1[3] = (int)((td1 >> 16) | ((unsigned)GBK << 16));    // td1 hi / tile_dim0
    g1[4] = (int)(unsigned)GBM;                            // tile_dim1=128
    g1[5] = (int)st0;                                      // dim0 stride lo32
    g1[6] = 0;                                             // stride hi / dim1str
    g1[7] = 0;
    v4i z4 = {0, 0, 0, 0};
    v8i z8 = {0, 0, 0, 0, 0, 0, 0, 0};
    // 6-arg form (this toolchain): (g0, g1, g2, g3, g4, cpol)
    __builtin_amdgcn_tensor_load_to_lds(g0, g1, z4, z4, z8, 0);
}
#endif

template <bool HAS_RES>
__global__ __launch_bounds__(256) void k_gemm_bf16(
    const unsigned short* __restrict__ A,
    const unsigned short* __restrict__ Bt,
    float* __restrict__ C,
    const float* __restrict__ res,
    int M, int N, int K) {
    __shared__ unsigned short As[2][GBM * GBK];   // 2 x 8 KB
    __shared__ unsigned short Bs[2][GBN * GBK];   // 2 x 8 KB

    const int tid  = threadIdx.x;
    const int lane = tid & 31;
    const int wave = tid >> 5;
    const int wm = wave >> 2;          // 0..1
    const int wn = wave & 3;           // 0..3
    const int laneHalf = lane >> 4;    // 0/1
    const int lane16   = lane & 15;

    const int block_m = blockIdx.y * GBM;
    const int block_n = blockIdx.x * GBN;

    v8f acc[4][2];
#pragma unroll
    for (int i = 0; i < 4; ++i)
#pragma unroll
        for (int j = 0; j < 2; ++j)
#pragma unroll
            for (int e = 0; e < 8; ++e) acc[i][j][e] = 0.f;

    const int nk = K / GBK;
    int buf = 0;

#ifdef USE_TDM
    // DMA the first K-slice; low 32 bits of a generic LDS pointer are the
    // LDS byte offset (flat-aperture truncation rule).
    if (wave == 0) {
        tdm_tile_load(A,  (unsigned)(size_t)&As[0][0], (unsigned)block_m, K, 0);
        tdm_tile_load(Bt, (unsigned)(size_t)&Bs[0][0], (unsigned)block_n, K, 0);
    }
    for (int ks = 0; ks < nk; ++ks) {
        if (wave == 0) __builtin_amdgcn_s_wait_tensorcnt(0);
        __syncthreads();                               // LDS[buf] ready
        if (wave == 0 && ks + 1 < nk) {                // DMA next slice
            const int k1 = (ks + 1) * GBK;
            tdm_tile_load(A,  (unsigned)(size_t)&As[buf ^ 1][0],
                          (unsigned)block_m, K, k1);
            tdm_tile_load(Bt, (unsigned)(size_t)&Bs[buf ^ 1][0],
                          (unsigned)block_n, K, k1);
        }
#else
    const int ldRow = tid >> 1;            // 0..127
    const int ldOfs = (tid & 1) * 16;      // element offset 0/16
    {   // preload slice 0 into LDS[0]
        uint4 av = *(const uint4*)(A + (size_t)(block_m + ldRow) * K + ldOfs);
        uint4 bv = *(const uint4*)(Bt + (size_t)(block_n + ldRow) * K + ldOfs);
        *(uint4*)&As[0][ldRow * GBK + ldOfs] = av;
        *(uint4*)&Bs[0][ldRow * GBK + ldOfs] = bv;
    }
    for (int ks = 0; ks < nk; ++ks) {
        __syncthreads();                               // LDS[buf] ready
        uint4 av = {}, bv = {};
        if (ks + 1 < nk) {                             // prefetch next slice
            const int k1 = (ks + 1) * GBK;
            av = *(const uint4*)(A + (size_t)(block_m + ldRow) * K + k1 + ldOfs);
            bv = *(const uint4*)(Bt + (size_t)(block_n + ldRow) * K + k1 + ldOfs);
        }
#endif
        // A fragments: ISA 16-bit A layout (lane m = lane16; elems 0..7 at
        // K = laneHalf*8, elems 8..15 at K = 16 + laneHalf*8)
        v16bf afrag[4];
#pragma unroll
        for (int i = 0; i < 4; ++i) {
            const unsigned short* p = &As[buf][(wm * 64 + i * 16 + lane16) * GBK];
            ((v8bf*)&afrag[i])[0] = *(const v8bf*)(p + laneHalf * 8);
            ((v8bf*)&afrag[i])[1] = *(const v8bf*)(p + 16 + laneHalf * 8);
        }
        // B fragments: lane col = lane16; elems 0..7 at K = laneHalf*8,
        // elems 8..15 at K = 16 + laneHalf*8 (same as A)
        v16bf bfrag[2];
#pragma unroll
        for (int j = 0; j < 2; ++j) {
            const unsigned short* p =
                &Bs[buf][(wn * 32 + j * 16 + lane16) * GBK];
            ((v8bf*)&bfrag[j])[0] = *(const v8bf*)(p + laneHalf * 8);
            ((v8bf*)&bfrag[j])[1] = *(const v8bf*)(p + 16 + laneHalf * 8);
        }
#pragma unroll
        for (int i = 0; i < 4; ++i)
#pragma unroll
            for (int j = 0; j < 2; ++j)
                acc[i][j] = __builtin_amdgcn_wmma_f32_16x16x32_bf16(
                    false, afrag[i], false, bfrag[j],
                    (short)0, acc[i][j], false, false);
#ifndef USE_TDM
        if (ks + 1 < nk) {
            *(uint4*)&As[buf ^ 1][ldRow * GBK + ldOfs] = av;
            *(uint4*)&Bs[buf ^ 1][ldRow * GBK + ldOfs] = bv;
        }
#endif
        buf ^= 1;
    }

    // store: VGPR r -> row r + 8*laneHalf, col lane16 (ISA 32-bit C/D layout)
    const int mBase = block_m + wm * 64 + laneHalf * 8;
    const int nBase = block_n + wn * 32 + lane16;
#pragma unroll
    for (int i = 0; i < 4; ++i) {
#pragma unroll
        for (int r = 0; r < 8; ++r) {
            const size_t rowOff = (size_t)(mBase + i * 16 + r) * N;
#pragma unroll
            for (int j = 0; j < 2; ++j) {
                const size_t idx = rowOff + nBase + j * 16;
                float v = acc[i][j][r];
                if (HAS_RES) v += res[idx];
                C[idx] = v;
            }
        }
    }
}

// ---------------------------------------------------------------------------
// Kernel 4: depthwise causal conv (4 taps) + SiLU.  xh = xz[:, 0:2048]
// ---------------------------------------------------------------------------
__global__ void k_conv_silu(const float* __restrict__ xz,
                            const float* __restrict__ cw,
                            const float* __restrict__ cb,
                            float* __restrict__ xc) {
    int idx = blockIdx.x * blockDim.x + threadIdx.x;   // over MROWS*DINNER
    if (idx >= MROWS * DINNER) return;
    int d   = idx & (DINNER - 1);
    int row = idx >> 11;
    int l   = row & (SEQ - 1);
    int b   = row >> 11;
    float acc = cb[d];
#pragma unroll
    for (int k = 0; k < DCONV; ++k) {
        int ll = l + k - (DCONV - 1);
        if (ll >= 0)
            acc += cw[d * DCONV + k] * xz[((size_t)(b * SEQ + ll)) * NXZ + d];
    }
    xc[idx] = acc / (1.f + expf(-acc));   // SiLU
}

// ---------------------------------------------------------------------------
// Kernel 5: xpj[row][j] = sum_d xc[row][d] * W_xp[d][j], j = 0..32
// ---------------------------------------------------------------------------
__global__ __launch_bounds__(256) void k_xproj(const float* __restrict__ xc,
                                               const float* __restrict__ Wxp,
                                               float* __restrict__ xpj) {
    __shared__ float sx[DINNER];
    __shared__ float ws[8];
    const int row = blockIdx.x, tid = threadIdx.x;
    const float* xr = xc + (size_t)row * DINNER;
    for (int i = tid; i < DINNER; i += 256) sx[i] = xr[i];
    __syncthreads();
    for (int j = 0; j < 2 * DSTATE + 1; ++j) {
        float p = 0.f;
        for (int d = tid; d < DINNER; d += 256) p += sx[d] * Wxp[d * 33 + j];
#pragma unroll
        for (int o = 16; o > 0; o >>= 1) p += __shfl_down(p, o, 32);
        if ((tid & 31) == 0) ws[tid >> 5] = p;
        __syncthreads();
        if (tid < 8) {
            p = ws[tid];
#pragma unroll
            for (int o = 4; o > 0; o >>= 1) p += __shfl_down(p, o, 8);
            if (tid == 0) xpj[(size_t)row * 33 + j] = p;
        }
        __syncthreads();
    }
}

// ---------------------------------------------------------------------------
// Kernel 6: selective scan. One thread per channel, h[16] in registers.
// ---------------------------------------------------------------------------
__global__ __launch_bounds__(64) void k_scan(const float* __restrict__ xpj,
                                             const float* __restrict__ xc,
                                             const float* __restrict__ Wdt,
                                             const float* __restrict__ bdt,
                                             const float* __restrict__ Alog,
                                             const float* __restrict__ Dp,
                                             float* __restrict__ y) {
    const int b = blockIdx.x >> 5;                       // 32 blocks per batch
    const int d = ((blockIdx.x & 31) << 6) + threadIdx.x;
    float A[DSTATE], h[DSTATE];
#pragma unroll
    for (int n = 0; n < DSTATE; ++n) {
        A[n] = -expf(Alog[d * DSTATE + n]);
        h[n] = 0.f;
    }
    const float wd = Wdt[d], bd = bdt[d], Dd = Dp[d];
    for (int l = 0; l < SEQ; ++l) {
        const size_t row = (size_t)b * SEQ + l;
        const float* pj = xpj + row * 33;      // lane-uniform (L2-resident)
        const float xv  = xc[row * DINNER + d];
        float t = pj[32] * wd + bd;
        float dt = (t > 20.f) ? t : log1pf(expf(t));   // softplus
        float acc = 0.f;
#pragma unroll
        for (int n = 0; n < DSTATE; ++n) {
            h[n] = expf(dt * A[n]) * h[n] + dt * pj[n] * xv;
            acc += h[n] * pj[16 + n];
        }
        y[row * DINNER + d] = acc + Dd * xv;
    }
}

// ---------------------------------------------------------------------------
// Kernel 7: gate yg = bf16(y * silu(z)),  z = xz[:, 2048:4096]
// ---------------------------------------------------------------------------
__global__ void k_gate(const float* __restrict__ y,
                       const float* __restrict__ xz,
                       unsigned short* __restrict__ yg) {
    int idx = blockIdx.x * blockDim.x + threadIdx.x;
    if (idx >= MROWS * DINNER) return;
    int d   = idx & (DINNER - 1);
    int row = idx >> 11;
    float z = xz[(size_t)row * NXZ + DINNER + d];
    float sz = z / (1.f + expf(-z));
    yg[idx] = f2bf(y[idx] * sz);
}

// ---------------------------------------------------------------------------
extern "C" void kernel_launch(void* const* d_in, const int* in_sizes, int n_in,
                              void* d_out, int out_size, void* d_ws, size_t ws_size,
                              hipStream_t stream) {
    const float* x      = (const float*)d_in[0];
    const float* ln_g   = (const float*)d_in[1];
    const float* ln_b   = (const float*)d_in[2];
    const float* W_in   = (const float*)d_in[3];
    const float* conv_w = (const float*)d_in[4];
    const float* conv_b = (const float*)d_in[5];
    const float* W_xp   = (const float*)d_in[6];
    const float* W_dt   = (const float*)d_in[7];
    const float* b_dt   = (const float*)d_in[8];
    const float* A_log  = (const float*)d_in[9];
    const float* Dp     = (const float*)d_in[10];
    const float* W_out  = (const float*)d_in[11];
    float* out = (float*)d_out;

    // workspace layout (~165 MB)
    char* base = (char*)d_ws;
    size_t off = 0;
    auto take = [&](size_t bytes) {
        char* p = base + off;
        off = (off + bytes + 255) & ~(size_t)255;
        return p;
    };
    unsigned short* xn_bf  = (unsigned short*)take((size_t)MROWS * DMODEL * 2);
    unsigned short* WinT   = (unsigned short*)take((size_t)NXZ * DMODEL * 2);
    unsigned short* WoutT  = (unsigned short*)take((size_t)DMODEL * DINNER * 2);
    float*          xz     = (float*)take((size_t)MROWS * NXZ * 4);
    float*          xc     = (float*)take((size_t)MROWS * DINNER * 4);
    float*          xpj    = (float*)take((size_t)MROWS * 33 * 4);
    float*          yb     = (float*)take((size_t)MROWS * DINNER * 4);
    unsigned short* yg     = (unsigned short*)take((size_t)MROWS * DINNER * 2);
    (void)ws_size;

    // 1. weight transpose+convert to bf16 [N][K]
    {
        int tot = DMODEL * NXZ;
        k_transpose_bf16<<<(tot + 255) / 256, 256, 0, stream>>>(W_in, WinT, DMODEL, NXZ);
    }
    {
        int tot = DINNER * DMODEL;
        k_transpose_bf16<<<(tot + 255) / 256, 256, 0, stream>>>(W_out, WoutT, DINNER, DMODEL);
    }
    // 2. layernorm
    k_layernorm_bf16<<<MROWS, 256, 0, stream>>>(x, ln_g, ln_b, xn_bf);
    // 3. xz = xn @ W_in  (4096x1024 @ 1024x4096)
    k_gemm_bf16<false><<<dim3(NXZ / GBN, MROWS / GBM), 256, 0, stream>>>(
        xn_bf, WinT, xz, nullptr, MROWS, NXZ, DMODEL);
    // 4. conv + silu
    {
        int tot = MROWS * DINNER;
        k_conv_silu<<<(tot + 255) / 256, 256, 0, stream>>>(xz, conv_w, conv_b, xc);
    }
    // 5. x-projection (B, C, dt_raw)
    k_xproj<<<MROWS, 256, 0, stream>>>(xc, W_xp, xpj);
    // 6. selective scan
    k_scan<<<BATCH * (DINNER / 64), 64, 0, stream>>>(xpj, xc, W_dt, b_dt, A_log, Dp, yb);
    // 7. gate
    {
        int tot = MROWS * DINNER;
        k_gate<<<(tot + 255) / 256, 256, 0, stream>>>(yb, xz, yg);
    }
    // 8. out = yg @ W_out + residual  (4096x2048 @ 2048x1024)
    k_gemm_bf16<true><<<dim3(DMODEL / GBN, MROWS / GBM), 256, 0, stream>>>(
        yg, WoutT, out, x, MROWS, DMODEL, DINNER);
}
